// TimeLSTMHyp_26603027431607
// MI455X (gfx1250) — matrix operand order repaired and checked by reading, not compiled, not run on hardware
//
#include <hip/hip_runtime.h>
#include <math.h>

#define B_ 64
#define S_ 128
#define H_ 512
#define I_ 512
#define MINN 1e-15f
#define BEPS 1e-5f
#define APAD 8   // halfs of row padding for A-stage LDS (rotates banks, 16B aligned)
#define WPAD 4   // floats of row padding for Wh LDS

typedef __attribute__((ext_vector_type(8)))  float    v8f;
typedef __attribute__((ext_vector_type(8)))  _Float16 v8h;
typedef __attribute__((ext_vector_type(16))) _Float16 v16h;

__device__ __forceinline__ float artanh_c(float x) {
    x = fminf(fmaxf(x, -1.f + BEPS), 1.f - BEPS);
    return 0.5f * logf((1.f + x) / (1.f - x));
}

// wave32 butterfly all-reduce (every lane gets the sum)
__device__ __forceinline__ float wred(float v) {
    #pragma unroll
    for (int off = 16; off > 0; off >>= 1) v += __shfl_xor(v, off, 32);
    return v;
}

// block reduction for the rescale kernel (blockDim multiple of 32, <=512)
__device__ __forceinline__ float block_reduce(float v, float* sbuf) {
    #pragma unroll
    for (int off = 16; off > 0; off >>= 1) v += __shfl_down(v, off, 32);
    const int lane = threadIdx.x & 31, w = threadIdx.x >> 5;
    const int nw = blockDim.x >> 5;
    __syncthreads();
    if (lane == 0) sbuf[w] = v;
    __syncthreads();
    float r = 0.f;
    if ((int)threadIdx.x < nw) r = sbuf[threadIdx.x];
    if (w == 0) {
        #pragma unroll
        for (int off = 8; off > 0; off >>= 1) r += __shfl_down(r, off, 32);
        if (lane == 0) sbuf[0] = r;
    }
    __syncthreads();
    return sbuf[0];
}

// ---------------------------------------------------------------------------
// One-time: repack weights to f16, B-operand friendly: W16[p*512+j][k]
//   p=0..3: W_all[j, p*512+k];  p=4: W_d[j,k]
// ---------------------------------------------------------------------------
__global__ void __launch_bounds__(256) conv_w_f16(const float* __restrict__ W_all,
                                                  const float* __restrict__ W_d,
                                                  _Float16* __restrict__ W16) {
    const int idx = blockIdx.x * 256 + threadIdx.x;   // < 2560*512
    const int col = idx >> 9;                          // p*512 + j
    const int k = idx & 511;
    const int p = col >> 9, j = col & 511;
    const float v = (p < 4) ? W_all[(size_t)j * (4 * H_) + p * H_ + k]
                            : W_d[(size_t)j * H_ + k];
    W16[idx] = (_Float16)v;
}

// ---------------------------------------------------------------------------
// Parallel precompute: C1[n, m] = sum_k X[n,k]*U_all[m,k]  (n = s*B+b)
// ---------------------------------------------------------------------------
__global__ void __launch_bounds__(256) ux_gemm_f16(const float* __restrict__ X,
                                                   const float* __restrict__ U,
                                                   float* __restrict__ C1) {
    const int wave = threadIdx.x >> 5;
    const int lane = threadIdx.x & 31;
    const int tile = blockIdx.x * 8 + wave;
    const int NT = (4 * H_) / 16;
    const int mtile = tile / NT, ntile = tile % NT;
    const int hi = (lane < 16) ? 0 : 1;
    const int l15 = lane & 15;

    const int nrow = mtile * 16 + l15;
    const int s = nrow >> 6, b = nrow & 63;
    const float* xrow = X + ((size_t)b * S_ + s) * I_;
    const float* urow = U + (size_t)(ntile * 16 + l15) * I_;

    v8f acc = {};
    for (int kk = 0; kk < I_; kk += 32) {
        v16h a, bm;
        #pragma unroll
        for (int j = 0; j < 16; ++j) {
            const int kh = (j < 8) ? j : (j + 8);
            a[j]  = (_Float16)xrow[kk + kh + hi * 8];
            bm[j] = (_Float16)urow[kk + kh + hi * 8];
        }
        acc = __builtin_amdgcn_wmma_f32_16x16x32_f16(false, a, false, bm,
                                                     (short)0, acc, false, false);
    }
    #pragma unroll
    for (int r = 0; r < 8; ++r)
        C1[(size_t)(mtile * 16 + r + hi * 8) * (4 * H_) + ntile * 16 + l15] = acc[r];
}

// In-place mobius_matvec rescale of C1 rows.
__global__ void __launch_bounds__(256) ux_rescale(const float* __restrict__ X,
                                                  float* __restrict__ C1) {
    __shared__ float sbuf[32];
    const int n = blockIdx.x;
    const int s = n >> 6, b = n & 63;
    const float* xrow = X + ((size_t)b * S_ + s) * I_;

    float xs = 0.f;
    for (int j = threadIdx.x; j < I_; j += blockDim.x) { float v = xrow[j]; xs += v * v; }
    const float xn = fmaxf(sqrtf(block_reduce(xs, sbuf)), MINN);
    const float at = artanh_c(xn);

    for (int g = 0; g < 4; ++g) {
        float* crow = C1 + (size_t)n * (4 * H_) + g * H_;
        const float v0 = crow[threadIdx.x];
        const float v1 = crow[threadIdx.x + 256];
        const float mxn = fmaxf(sqrtf(block_reduce(v0 * v0 + v1 * v1, sbuf)), MINN);
        const float sc = tanhf(mxn / xn * at) / mxn;
        crow[threadIdx.x]       = v0 * sc;
        crow[threadIdx.x + 256] = v1 * sc;
    }
}

// ---------------------------------------------------------------------------
// Persistent scan kernel: 4 blocks x 512 threads. Block owns 16 batch rows
// (one WMMA M-tile); wave w == row w for pointwise (all reductions wave32
// shuffles). Per part: GEMM f16-WMMA into LDS -> barrier -> pointwise.
// ---------------------------------------------------------------------------
__global__ void __launch_bounds__(512) hyp_lstm_scan(
    const float* __restrict__ C1, const _Float16* __restrict__ W16,
    const float* __restrict__ ts, const float* __restrict__ h0,
    const float* __restrict__ c0, float* __restrict__ outputs,
    float* __restrict__ h_f, float* __restrict__ c_f) {
    __shared__ _Float16 AB[16][H_ + APAD];   // staged A operand (h or c), f16
    __shared__ float WhL[16][H_ + WPAD];     // current-part GEMM result

    const int w = threadIdx.x >> 5;          // wave id == row-in-block
    const int l = threadIdx.x & 31;
    const int l15 = l & 15;
    const int hi = (l < 16) ? 0 : 1;
    const int b = blockIdx.x * 16 + w;       // global batch row (pointwise)

    float hreg[16], creg[16];
    #pragma unroll
    for (int q = 0; q < 16; ++q) {
        hreg[q] = h0[(size_t)b * H_ + l + 32 * q];
        creg[q] = c0[(size_t)b * H_ + l + 32 * q];
    }

    for (int s = 0; s < S_; ++s) {
        // state norms (before hreg/creg are overwritten)
        float h2l = 0.f, c2l = 0.f;
        #pragma unroll
        for (int q = 0; q < 16; ++q) { h2l += hreg[q] * hreg[q]; c2l += creg[q] * creg[q]; }
        const float h2 = wred(h2l), c2 = wred(c2l);
        const float hn = fmaxf(sqrtf(h2), MINN), cn = fmaxf(sqrtf(c2), MINN);
        const float ath = artanh_c(hn);
        const float t = ts[(size_t)b * S_ + s];

        // GEMM of one part into WhL (A staged in AB). 2 N-tiles per wave.
        auto gemm_part = [&](int p) {
            #pragma unroll 1
            for (int tt = 0; tt < 2; ++tt) {
                const int jt = 2 * w + tt;
                const int j = jt * 16 + l15;
                const _Float16* Brow = W16 + (size_t)(p * H_ + j) * H_;
                __builtin_prefetch(Brow + H_, 0, 1);       // next column's panel
                v8f acc = {};
                for (int kk = 0; kk < H_; kk += 32) {
                    const v8h a0 = *(const v8h*)&AB[l15][kk + hi * 8];
                    const v8h a1 = *(const v8h*)&AB[l15][kk + 16 + hi * 8];
                    const v8h b0 = *(const v8h*)&Brow[kk + hi * 8];
                    const v8h b1 = *(const v8h*)&Brow[kk + 16 + hi * 8];
                    v16h a, bm;
                    #pragma unroll
                    for (int jj = 0; jj < 8; ++jj) {
                        a[jj] = a0[jj]; a[jj + 8] = a1[jj];
                        bm[jj] = b0[jj]; bm[jj + 8] = b1[jj];
                    }
                    acc = __builtin_amdgcn_wmma_f32_16x16x32_f16(false, a, false, bm,
                                                                 (short)0, acc, false, false);
                }
                #pragma unroll
                for (int r8 = 0; r8 < 8; ++r8)
                    WhL[r8 + hi * 8][jt * 16 + l15] = acc[r8];
            }
        };

        // sigmoid(logmap0(mobius_add(mv(W_g,h), ux_g))) for gate g -> dst
        auto gate_compute = [&](int g, float* dst) {
            float wv[16], ux[16];
            float accw = 0.f;
            #pragma unroll
            for (int q = 0; q < 16; ++q) { wv[q] = WhL[w][l + 32 * q]; accw += wv[q] * wv[q]; }
            const float wh2 = wred(accw);
            const float whn = fmaxf(sqrtf(wh2), MINN);
            const float sg = tanhf(whn / hn * ath) / whn;
            const float mv2 = sg * sg * wh2;
            const float* uxr = C1 + ((size_t)(s * B_ + b)) * (4 * H_) + g * H_;
            float accu = 0.f, accxy = 0.f;
            #pragma unroll
            for (int q = 0; q < 16; ++q) {
                ux[q] = uxr[l + 32 * q];
                accu += ux[q] * ux[q];
                accxy += sg * wv[q] * ux[q];
            }
            const float ux2 = wred(accu), xyg = wred(accxy);
            const float dg = fmaxf(1.f + 2.f * xyg + mv2 * ux2, MINN);
            const float am = (1.f + 2.f * xyg + ux2) * sg / dg;
            const float bm = (1.f - mv2) / dg;
            float accm = 0.f;
            #pragma unroll
            for (int q = 0; q < 16; ++q) { wv[q] = am * wv[q] + bm * ux[q]; accm += wv[q] * wv[q]; }
            const float mn = fmaxf(sqrtf(wred(accm)), MINN);
            const float ls = artanh_c(mn) / mn;
            #pragma unroll
            for (int q = 0; q < 16; ++q) dst[q] = 1.f / (1.f + expf(-ls * wv[q]));
        };

        // ===== part 4: Mc = c @ W_d^T, time-decay branch =====
        __syncthreads();
        #pragma unroll
        for (int q = 0; q < 16; ++q) AB[w][l + 32 * q] = (_Float16)creg[q];
        __syncthreads();
        gemm_part(4);
        __syncthreads();

        float u[16], cadj[16];
        float acc = 0.f;
        #pragma unroll
        for (int q = 0; q < 16; ++q) { u[q] = WhL[w][l + 32 * q]; acc += u[q] * u[q]; }
        const float Mcn = fmaxf(sqrtf(wred(acc)), MINN);
        const float sd = tanhf(Mcn / cn * artanh_c(cn)) / Mcn;
        const float mvdn = fmaxf(fabsf(sd) * Mcn, MINN);
        const float lsc = artanh_c(mvdn) / mvdn * sd;       // logmap0 as scalar on Mc
        acc = 0.f;
        #pragma unroll
        for (int q = 0; q < 16; ++q) { u[q] = tanhf(lsc * u[q]); acc += u[q] * u[q]; }
        const float un = fmaxf(sqrtf(wred(acc)), MINN);
        const float se = tanhf(un) / un;                     // cs1 = se*u
        const float cs1n = fmaxf(tanhf(un), MINN), cs1sq = cs1n * cs1n;
        const float xnb = fmaxf(fabsf(t) * sqrtf((float)H_), MINN);
        const float wxnb = fmaxf(fabsf(t) * cs1n, MINN);
        const float s2 = tanhf(wxnb / xnb * artanh_c(xnb)) / wxnb;
        const float cs2sc = s2 * t * se;                     // cs2 = cs2sc*u
        const float cs2sq = (s2 * t * cs1n) * (s2 * t * cs1n);
        acc = 0.f;
        #pragma unroll
        for (int q = 0; q < 16; ++q) acc += (se * u[q]) * creg[q];
        const float xy = -wred(acc);
        const float den = fmaxf(1.f + 2.f * xy + cs1sq * c2, MINN);
        const float a1 = -(1.f + 2.f * xy + c2) * se / den;
        const float b1 = (1.f - cs1sq) / den;                // cl = a1*u + b1*c
        acc = 0.f; float acc2 = 0.f;
        #pragma unroll
        for (int q = 0; q < 16; ++q) {
            cadj[q] = a1 * u[q] + b1 * creg[q];              // cl for now
            acc += cadj[q] * cadj[q];
            acc2 += cadj[q] * (cs2sc * u[q]);
        }
        const float cl2 = wred(acc), xy2 = wred(acc2);
        const float den2 = fmaxf(1.f + 2.f * xy2 + cl2 * cs2sq, MINN);
        const float a2 = (1.f + 2.f * xy2 + cs2sq) / den2;
        const float b2 = (1.f - cl2) * cs2sc / den2;
        acc = 0.f;
        #pragma unroll
        for (int q = 0; q < 16; ++q) { cadj[q] = a2 * cadj[q] + b2 * u[q]; acc += cadj[q] * cadj[q]; }
        const float cadj2 = wred(acc);
        __syncthreads();

        // ===== gates (A = h) =====
        #pragma unroll
        for (int q = 0; q < 16; ++q) AB[w][l + 32 * q] = (_Float16)hreg[q];
        __syncthreads();

        float ga[16], gb[16], pwi[16], pwf[16];

        gemm_part(1); __syncthreads();       // input gate
        gate_compute(1, ga); __syncthreads();
        gemm_part(3); __syncthreads();       // candidate
        gate_compute(3, gb); __syncthreads();
        // pwi = mobius_pw_mul(i, c_tmp)
        acc = 0.f; acc2 = 0.f;
        #pragma unroll
        for (int q = 0; q < 16; ++q) {
            pwi[q] = ga[q] * gb[q];
            acc += gb[q] * gb[q];
            acc2 += pwi[q] * pwi[q];
        }
        const float ctn = fmaxf(sqrtf(wred(acc)), MINN);
        const float wxin = fmaxf(sqrtf(wred(acc2)), MINN);
        const float ti = tanhf(wxin / ctn * artanh_c(ctn));
        #pragma unroll
        for (int q = 0; q < 16; ++q) pwi[q] *= ti / wxin;

        gemm_part(0); __syncthreads();       // forget gate
        gate_compute(0, ga); __syncthreads();
        // pwf = mobius_pw_mul(f, c_adj)
        const float cadjn = fmaxf(sqrtf(cadj2), MINN);
        acc = 0.f;
        #pragma unroll
        for (int q = 0; q < 16; ++q) { pwf[q] = ga[q] * cadj[q]; acc += pwf[q] * pwf[q]; }
        const float wxfn = fmaxf(sqrtf(wred(acc)), MINN);
        const float tf = tanhf(wxfn / cadjn * artanh_c(cadjn));
        #pragma unroll
        for (int q = 0; q < 16; ++q) pwf[q] *= tf / wxfn;
        // c_new = mobius_add(pwi, pwf)
        acc = 0.f;
        #pragma unroll
        for (int q = 0; q < 16; ++q) acc += pwi[q] * pwf[q];
        const float xy3 = wred(acc);
        const float den3 = fmaxf(1.f + 2.f * xy3 + ti * ti * tf * tf, MINN);
        #pragma unroll
        for (int q = 0; q < 16; ++q)
            creg[q] = ((1.f + 2.f * xy3 + tf * tf) * pwi[q] + (1.f - ti * ti) * pwf[q]) / den3;

        gemm_part(2); __syncthreads();       // output gate
        gate_compute(2, ga); __syncthreads();
        // h_new = mobius_pw_mul(o, expmap0(tanh(c_new)))
        acc = 0.f;
        #pragma unroll
        for (int q = 0; q < 16; ++q) { u[q] = tanhf(creg[q]); acc += u[q] * u[q]; }
        const float en = fmaxf(sqrtf(wred(acc)), MINN);
        const float ems = tanhf(en) / en;
        const float emn = fmaxf(tanhf(en), MINN);
        acc = 0.f;
        #pragma unroll
        for (int q = 0; q < 16; ++q) { u[q] = ga[q] * ems * u[q]; acc += u[q] * u[q]; }
        const float wxon = fmaxf(sqrtf(wred(acc)), MINN);
        const float to = tanhf(wxon / emn * artanh_c(emn));
        #pragma unroll
        for (int q = 0; q < 16; ++q) {
            hreg[q] = to / wxon * u[q];
            outputs[((size_t)b * S_ + s) * H_ + l + 32 * q] = ga[q];
        }
    }

    #pragma unroll
    for (int q = 0; q < 16; ++q) {
        h_f[(size_t)b * H_ + l + 32 * q] = hreg[q];
        c_f[(size_t)b * H_ + l + 32 * q] = creg[q];
    }
}

extern "C" void kernel_launch(void* const* d_in, const int* in_sizes, int n_in,
                              void* d_out, int out_size, void* d_ws, size_t ws_size,
                              hipStream_t stream) {
    const float* inputs = (const float*)d_in[0];   // [B,S,I]
    const float* tsamp  = (const float*)d_in[1];   // [B,S]
    const float* h0     = (const float*)d_in[2];   // [B,H]
    const float* c0     = (const float*)d_in[3];   // [B,H]
    const float* W_all  = (const float*)d_in[4];   // [H,4H]
    const float* U_all  = (const float*)d_in[5];   // [4H,I]
    const float* W_d    = (const float*)d_in[6];   // [H,H]

    float* C1 = (float*)d_ws;                            // [S*B, 4H] (64 MB)
    _Float16* W16 = (_Float16*)(C1 + (size_t)S_ * B_ * 4 * H_);  // [5*H, H] f16

    float* outputs = (float*)d_out;                      // [B,S,H]
    float* h_f = outputs + (size_t)B_ * S_ * H_;
    float* c_f = h_f + (size_t)B_ * H_;

    conv_w_f16<<<(5 * H_ * H_) / 256, 256, 0, stream>>>(W_all, W_d, W16);
    ux_gemm_f16<<<(S_ * B_ / 16) * (4 * H_ / 16) / 8, 256, 0, stream>>>(inputs, U_all, C1);
    ux_rescale<<<S_ * B_, 256, 0, stream>>>(inputs, C1);
    hyp_lstm_scan<<<4, 512, 0, stream>>>(C1, W16, tsamp, h0, c0, outputs, h_f, c_f);
}
